// ViT_44805098832569
// MI455X (gfx1250) — matrix operand order repaired and checked by reading, not compiled, not run on hardware
//
#include <hip/hip_runtime.h>
#include <hip/hip_bf16.h>
#include <math.h>

// ---------------- types ----------------
typedef unsigned short u16;
typedef __attribute__((ext_vector_type(16))) __bf16 v16bf;
typedef __attribute__((ext_vector_type(8)))  __bf16 v8bf;
typedef __attribute__((ext_vector_type(8)))  float  v8f;

// ---------------- model dims ----------------
#define NLAYER 12
#define NHEAD  16
#define DHEAD  32
#define DMODEL 512
#define DFF    2048
#define SEQ    17
#define NCLS   10
#define BATCH  2048
#define PDIM   192

__device__ __forceinline__ u16 f2bf(float f) {
  unsigned u = __float_as_uint(f);
  unsigned r = (u + 0x7FFFu + ((u >> 16) & 1u)) >> 16;   // RNE
  return (u16)r;
}

// ============================================================
// bf16 WMMA GEMM:  D = epilogue(A(MxK) * B(KxN) + bias [+C])
// B is PRE-TRANSPOSED (Bt = N x K row-major). K stepped by 64.
// Staging uses CDNA5 GLOBAL_LOAD_ASYNC_TO_LDS_B128 (ASYNCcnt) into a
// double-buffered LDS tile: next slice's loads are in flight across the
// entire WMMA phase, no VGPR transit, one barrier per K-step.
// Block = 256 thr = 8 waves (4M x 2N); block tile 256x64; wave tile 64x32.
// M multiple of 256, K multiple of 64 (true for all uses: 192/512/2048).
// Bt must have >= ceil64(N) rows (classifier weight padded with zeros).
// row_remap=1 maps output row r -> r + r/16 (16-token -> 17-token rows).
// ============================================================
enum { EPI_F32 = 0, EPI_LEAKY_BF16 = 1, EPI_GELU_BF16 = 2, EPI_RESID_F32 = 3 };

template<int EPI>
__global__ __launch_bounds__(256)
void gemm_bf16_kernel(const u16* __restrict__ A, const u16* __restrict__ Bt,
                      const float* __restrict__ bias, const float* __restrict__ Cres,
                      void* __restrict__ Dout, int M, int N, int K, int row_remap)
{
  constexpr int AP = 72;                 // 64 + 8 pad; row pitch 144B (16B aligned)
  constexpr int BP = 72;
  __shared__ __attribute__((aligned(16))) u16 lA[2][256 * AP];   // 2 x 36.9 KB
  __shared__ __attribute__((aligned(16))) u16 lBt[2][64 * BP];   // 2 x  9.2 KB

  const int t    = threadIdx.x;
  const int lane = t & 31;
  const int wave = t >> 5;
  const int wm   = wave >> 1;            // 0..3 -> M offset (x64)
  const int wn   = wave & 1;             // 0..1 -> N offset (x32)
  const int mBase = blockIdx.y * 256;
  const int nBase = blockIdx.x * 64;
  const int lhalf = lane >> 4;           // lane half (0/1)
  const int mr    = lane & 15;

  // staging coordinates: each row is 8 chunks of 8 elems; thread t covers
  // K-offset (t&7)*8, A rows (t>>3)+i*32 (i<8), Bt rows (t>>3)+i*32 (i<2)
  const int sRow = t >> 3;               // 0..31
  const int sKc  = (t & 7) << 3;         // 0..56

  const u16* aP = A  + (size_t)(mBase + sRow) * K + sKc;
  const u16* bP = Bt + (size_t)(nBase + sRow) * K + sKc;
  const size_t rowStep32 = (size_t)32 * K;

  v8f acc[4][2] = {};

  // issue one 64-wide K-slice's async LDS loads for buffer `buf`
  auto stage = [&](int buf, const u16* aQ, const u16* bQ) {
#pragma unroll
    for (int i = 0; i < 8; ++i) {
      unsigned dst = (unsigned)(uintptr_t)&lA[buf][(sRow + i * 32) * AP + sKc];
      const u16* src = aQ + (size_t)i * rowStep32;
      asm volatile("global_load_async_to_lds_b128 %0, %1, off"
                   :: "v"(dst), "v"(src) : "memory");
    }
#pragma unroll
    for (int i = 0; i < 2; ++i) {
      unsigned dst = (unsigned)(uintptr_t)&lBt[buf][(sRow + i * 32) * BP + sKc];
      const u16* src = bQ + (size_t)i * rowStep32;
      asm volatile("global_load_async_to_lds_b128 %0, %1, off"
                   :: "v"(dst), "v"(src) : "memory");
    }
  };

  stage(0, aP, bP);
  int cur = 0;

  for (int k0 = 0; k0 < K; k0 += 64) {
    asm volatile("s_wait_asynccnt 0x0" ::: "memory");   // this wave's LDS writes landed
    __syncthreads();                                    // everyone's writes landed / reads done
    if (k0 + 64 < K) {
      aP += 64;
      bP += 64;
      stage(cur ^ 1, aP, bP);                           // prefetch next slice (other buffer)
    }

    const u16* lAc = lA[cur];
    const u16* lBc = lBt[cur];

    // ---- two 32-wide WMMA sub-steps ----
#pragma unroll
    for (int ks = 0; ks < 64; ks += 32) {
      // fragments per ISA 7.12.2 layouts:
      // A lane m=L&15: K chunks at ks+{8*half..+8} and ks+{16+8*half..+8}
      // B lane n=L&15: K chunks at ks+{8*half..+8} and ks+{16+8*half..+8}
      v16bf af[4], bfr[2];
#pragma unroll
      for (int j = 0; j < 2; ++j) {
        int col = wn * 32 + j * 16 + mr;
        v8bf lo = *reinterpret_cast<const v8bf*>(&lBc[col * BP + ks + lhalf * 8]);
        v8bf hi = *reinterpret_cast<const v8bf*>(&lBc[col * BP + ks + 16 + lhalf * 8]);
        bfr[j] = __builtin_shufflevector(lo, hi, 0,1,2,3,4,5,6,7,8,9,10,11,12,13,14,15);
      }
#pragma unroll
      for (int i = 0; i < 4; ++i) {
        int row = wm * 64 + i * 16 + mr;
        v8bf lo = *reinterpret_cast<const v8bf*>(&lAc[row * AP + ks + lhalf * 8]);
        v8bf hi = *reinterpret_cast<const v8bf*>(&lAc[row * AP + ks + 16 + lhalf * 8]);
        af[i] = __builtin_shufflevector(lo, hi, 0,1,2,3,4,5,6,7,8,9,10,11,12,13,14,15);
      }
#pragma unroll
      for (int i = 0; i < 4; ++i)
#pragma unroll
        for (int j = 0; j < 2; ++j)
          acc[i][j] = __builtin_amdgcn_wmma_f32_16x16x32_bf16(
              false, af[i], false, bfr[j], (short)0, acc[i][j], false, false);
    }
    cur ^= 1;
  }

  // ---- epilogue ----
#pragma unroll
  for (int i = 0; i < 4; ++i) {
#pragma unroll
    for (int j = 0; j < 2; ++j) {
      int col = nBase + wn * 32 + j * 16 + mr;
      if (col >= N) continue;            // only trims for the N=10 classifier
      float bv = bias[col];
#pragma unroll
      for (int r = 0; r < 8; ++r) {
        int row  = mBase + wm * 64 + i * 16 + lhalf * 8 + r;
        int orow = row_remap ? (row + (row >> 4)) : row;
        size_t oi = (size_t)orow * N + col;
        float v = acc[i][j][r] + bv;
        if (EPI == EPI_F32) {
          ((float*)Dout)[oi] = v;
        } else if (EPI == EPI_RESID_F32) {
          ((float*)Dout)[oi] = v + Cres[oi];
        } else if (EPI == EPI_LEAKY_BF16) {
          ((u16*)Dout)[oi] = f2bf(v > 0.f ? v : 0.1f * v);
        } else { // EPI_GELU_BF16 (exact gelu)
          ((u16*)Dout)[oi] = f2bf(0.5f * v * (1.f + erff(v * 0.70710678118654752f)));
        }
      }
    }
  }
}

// ============================================================
// f32 -> bf16 TRANSPOSING convert: src (K x N) -> dst (Npad x K), rows
// n >= N zero-filled. One-time pass over the weights.
// ============================================================
__global__ __launch_bounds__(256)
void transpose_convert_kernel(const float* __restrict__ src, u16* __restrict__ dst,
                              int N, int K, int Npad) {
  int idx = blockIdx.x * 256 + threadIdx.x;
  if (idx >= Npad * K) return;
  int n = idx / K;
  int k = idx - n * K;
  dst[idx] = (n < N) ? f2bf(src[(size_t)k * N + n]) : (u16)0;
}

// ============================================================
// Conv: 3x3x16x16 kernel, pad (7,8),(7,8), NCHW, out (B,3,32,32)
// ============================================================
__global__ __launch_bounds__(256)
void conv_kernel(const float* __restrict__ X, const float* __restrict__ W,
                 const float* __restrict__ Bc, float* __restrict__ out, int total) {
  int idx = blockIdx.x * 256 + threadIdx.x;
  if (idx >= total) return;
  int x0 = idx & 31;
  int y0 = (idx >> 5) & 31;
  int o  = (idx >> 10) % 3;
  int b  = idx / (3 * 1024);
  float acc = Bc[o];
  for (int i = 0; i < 3; ++i) {
    const float* xi = X + (size_t)(b * 3 + i) * 1024;
    const float* wi = W + (size_t)(o * 3 + i) * 256;
    for (int ky = 0; ky < 16; ++ky) {
      int yy = y0 + ky - 7;
      if ((unsigned)yy >= 32u) continue;
#pragma unroll
      for (int kx = 0; kx < 16; ++kx) {
        int xx = x0 + kx - 7;
        if ((unsigned)xx >= 32u) continue;
        acc = fmaf(xi[yy * 32 + xx], wi[ky * 16 + kx], acc);
      }
    }
  }
  out[idx] = acc;
}

// ============================================================
// Patchify (B,3,32,32) -> (B*16, 192) bf16
// ============================================================
__global__ __launch_bounds__(256)
void patchify_kernel(const float* __restrict__ xc, u16* __restrict__ p, int total) {
  int idx = blockIdx.x * 256 + threadIdx.x;
  if (idx >= total) return;
  int f    = idx % PDIM;
  int rest = idx / PDIM;
  int pi   = rest & 15;
  int b    = rest >> 4;
  int py = pi >> 2, px = pi & 3;
  int c  = f >> 6;
  int r2 = f & 63;
  int dy = r2 >> 3, dx = r2 & 7;
  float v = xc[((size_t)(b * 3 + c) * 32 + (py * 8 + dy)) * 32 + (px * 8 + dx)];
  p[idx] = f2bf(v);
}

// ============================================================
// Assemble h: add pos_emb, write cls row (token 16)
// ============================================================
__global__ __launch_bounds__(256)
void assemble_kernel(float* __restrict__ h, const float* __restrict__ cls_tok,
                     const float* __restrict__ pos, int total) {
  int idx = blockIdx.x * 256 + threadIdx.x;
  if (idx >= total) return;
  int d   = idx & 511;
  int tok = (idx >> 9) % SEQ;
  if (tok == 16) h[idx] = cls_tok[d] + pos[16 * DMODEL + d];
  else           h[idx] += pos[tok * DMODEL + d];
}

// ============================================================
// LayerNorm over D=512, one wave per row, out bf16.
// rowmap=1: source row r -> r*17+16 (cls extraction for final LN)
// ============================================================
__global__ __launch_bounds__(256)
void ln_kernel(const float* __restrict__ X, const float* __restrict__ s,
               const float* __restrict__ b, u16* __restrict__ out,
               int nrows, int rowmap) {
  int wave = threadIdx.x >> 5, lane = threadIdx.x & 31;
  int r = blockIdx.x * 8 + wave;
  if (r >= nrows) return;
  size_t srow = rowmap ? ((size_t)r * SEQ + 16) : (size_t)r;
  const float* x = X + srow * DMODEL;
  float v[16];
  float sum = 0.f, sq = 0.f;
#pragma unroll
  for (int j = 0; j < 16; ++j) {
    v[j] = x[lane + j * 32];
    sum += v[j];
    sq  = fmaf(v[j], v[j], sq);
  }
  for (int off = 16; off; off >>= 1) {
    sum += __shfl_xor(sum, off, 32);
    sq  += __shfl_xor(sq, off, 32);
  }
  float mean = sum * (1.f / 512.f);
  float var  = sq * (1.f / 512.f) - mean * mean;
  float rs   = rsqrtf(var + 1e-5f);
  u16* o = out + (size_t)r * DMODEL;
#pragma unroll
  for (int j = 0; j < 16; ++j) {
    int d = lane + j * 32;
    o[d] = f2bf((v[j] - mean) * rs * s[d] + b[d]);
  }
}

// ============================================================
// Attention: one wave per (batch, head). S=17, DH=32 (= lane dim).
// qkv f32 (B*17, 1536), out bf16 (B*17, 512) with heads concatenated.
// ============================================================
__global__ __launch_bounds__(256)
void attn_kernel(const float* __restrict__ qkv, u16* __restrict__ out) {
  int lane = threadIdx.x & 31;
  int wv   = threadIdx.x >> 5;
  int bh   = blockIdx.x * 8 + wv;            // 0 .. B*H-1
  int b = bh >> 4;
  int hh = bh & 15;
  const float* base = qkv + (size_t)b * SEQ * (3 * DMODEL);

  // lane t holds K row t (t>=17 clamped, masked later)
  float krow[32];
  {
    int tr = lane < SEQ ? lane : SEQ - 1;
    const float* kp = base + (size_t)tr * (3 * DMODEL) + DMODEL + hh * DHEAD;
#pragma unroll
    for (int d = 0; d < 32; ++d) krow[d] = kp[d];
  }
  // lane d holds V column d
  float vcol[SEQ];
#pragma unroll
  for (int tt = 0; tt < SEQ; ++tt)
    vcol[tt] = base[(size_t)tt * (3 * DMODEL) + 2 * DMODEL + hh * DHEAD + lane];

  const float scale = 0.17677669529663687f;  // 1/sqrt(32)
  for (int s0 = 0; s0 < SEQ; ++s0) {
    float qv = base[(size_t)s0 * (3 * DMODEL) + hh * DHEAD + lane];
    float sc = 0.f;
#pragma unroll
    for (int d = 0; d < 32; ++d) sc = fmaf(__shfl(qv, d, 32), krow[d], sc);
    sc *= scale;
    if (lane >= SEQ) sc = -INFINITY;
    float mx = sc;
    for (int off = 16; off; off >>= 1) mx = fmaxf(mx, __shfl_xor(mx, off, 32));
    float e = (lane < SEQ) ? expf(sc - mx) : 0.f;
    float se = e;
    for (int off = 16; off; off >>= 1) se += __shfl_xor(se, off, 32);
    float p = e / se;
    float ov = 0.f;
#pragma unroll
    for (int tt = 0; tt < SEQ; ++tt) ov = fmaf(__shfl(p, tt, 32), vcol[tt], ov);
    out[(size_t)(b * SEQ + s0) * DMODEL + hh * DHEAD + lane] = f2bf(ov);
  }
}

// ============================================================
// Host-side orchestration
// ============================================================
static inline int ceil_div(int a, int b) { return (a + b - 1) / b; }

extern "C" void kernel_launch(void* const* d_in, const int* in_sizes, int n_in,
                              void* d_out, int out_size, void* d_ws, size_t ws_size,
                              hipStream_t stream) {
  (void)in_sizes; (void)n_in; (void)out_size; (void)ws_size;

  const float* x       = (const float*)d_in[0];
  const float* conv_w  = (const float*)d_in[1];
  const float* conv_b  = (const float*)d_in[2];
  const float* h_w1    = (const float*)d_in[3];
  const float* h_b1    = (const float*)d_in[4];
  const float* h_w2    = (const float*)d_in[5];
  const float* h_b2    = (const float*)d_in[6];
  const float* cls_tok = (const float*)d_in[7];
  const float* pos_emb = (const float*)d_in[8];
  const float* ln1_s   = (const float*)d_in[9];
  const float* ln1_b   = (const float*)d_in[10];
  const float* qkv_w   = (const float*)d_in[11];
  const float* qkv_b   = (const float*)d_in[12];
  const float* proj_w  = (const float*)d_in[13];
  const float* proj_b  = (const float*)d_in[14];
  const float* ln2_s   = (const float*)d_in[15];
  const float* ln2_b   = (const float*)d_in[16];
  const float* mlp_w1  = (const float*)d_in[17];
  const float* mlp_b1  = (const float*)d_in[18];
  const float* mlp_w2  = (const float*)d_in[19];
  const float* mlp_b2  = (const float*)d_in[20];
  const float* fin_s   = (const float*)d_in[21];
  const float* fin_b   = (const float*)d_in[22];
  const float* cls_w   = (const float*)d_in[23];
  const float* cls_b   = (const float*)d_in[24];

  // ---- workspace layout ----
  char* ws = (char*)d_ws;
  size_t off = 0;
  auto walloc = [&](size_t bytes) -> void* {
    void* p = ws + off;
    off += (bytes + 255) & ~(size_t)255;
    return p;
  };
  // transposed bf16 weights (Bt = N x K row-major)
  u16* h_w1T   = (u16*)walloc((size_t)DMODEL * PDIM * 2);
  u16* h_w2T   = (u16*)walloc((size_t)DMODEL * DMODEL * 2);
  u16* qkv_wT  = (u16*)walloc((size_t)NLAYER * 3 * DMODEL * DMODEL * 2);
  u16* proj_wT = (u16*)walloc((size_t)NLAYER * DMODEL * DMODEL * 2);
  u16* mlp_w1T = (u16*)walloc((size_t)NLAYER * DFF * DMODEL * 2);
  u16* mlp_w2T = (u16*)walloc((size_t)NLAYER * DMODEL * DFF * 2);
  u16* cls_wT  = (u16*)walloc((size_t)64 * DMODEL * 2);                  // padded to 64 rows
  float* hF32  = (float*)walloc((size_t)BATCH * SEQ * DMODEL * 4);
  u16* nb      = (u16*)walloc((size_t)BATCH * SEQ * DMODEL * 2);         // LN out / attn out / cls in
  char* big    = (char*)walloc((size_t)BATCH * SEQ * 3 * DMODEL * 4);    // qkv f32 | mlp hidden bf16 | patch temps
  float* qkvF  = (float*)big;
  u16*   hidB  = (u16*)big;
  float* xcF   = (float*)big;                                            // conv out (start of pipeline only)
  u16*   pB    = (u16*)(big + (size_t)BATCH * 3 * 32 * 32 * 4);          // patches bf16
  u16*   hid1B = (u16*)(big + (size_t)BATCH * 3 * 32 * 32 * 4
                            + (size_t)BATCH * 16 * PDIM * 2);            // patch hidden bf16

  const int M16 = BATCH * 16;   // 32768
  const int M17 = BATCH * SEQ;  // 34816

  // ---- 1. transpose-convert all weights to bf16 (once per launch) ----
  auto tconv = [&](const float* s, u16* d, int N, int K, int Npad) {
    transpose_convert_kernel<<<ceil_div(Npad * K, 256), 256, 0, stream>>>(s, d, N, K, Npad);
  };
  tconv(h_w1, h_w1T, DMODEL, PDIM, DMODEL);
  tconv(h_w2, h_w2T, DMODEL, DMODEL, DMODEL);
  for (int l = 0; l < NLAYER; ++l) {
    tconv(qkv_w  + (size_t)l * DMODEL * 3 * DMODEL, qkv_wT  + (size_t)l * 3 * DMODEL * DMODEL,
          3 * DMODEL, DMODEL, 3 * DMODEL);
    tconv(proj_w + (size_t)l * DMODEL * DMODEL,     proj_wT + (size_t)l * DMODEL * DMODEL,
          DMODEL, DMODEL, DMODEL);
    tconv(mlp_w1 + (size_t)l * DMODEL * DFF,        mlp_w1T + (size_t)l * DFF * DMODEL,
          DFF, DMODEL, DFF);
    tconv(mlp_w2 + (size_t)l * DFF * DMODEL,        mlp_w2T + (size_t)l * DMODEL * DFF,
          DMODEL, DFF, DMODEL);
  }
  tconv(cls_w, cls_wT, NCLS, DMODEL, 64);

  // ---- 2. conv + patchify ----
  {
    int total = BATCH * 3 * 32 * 32;
    conv_kernel<<<ceil_div(total, 256), 256, 0, stream>>>(x, conv_w, conv_b, xcF, total);
    int ptotal = M16 * PDIM;
    patchify_kernel<<<ceil_div(ptotal, 256), 256, 0, stream>>>(xcF, pB, ptotal);
  }

  // ---- 3. patch MLP ----
  gemm_bf16_kernel<EPI_LEAKY_BF16><<<dim3(DMODEL / 64, M16 / 256), 256, 0, stream>>>(
      pB, h_w1T, h_b1, nullptr, hid1B, M16, DMODEL, PDIM, 0);
  gemm_bf16_kernel<EPI_F32><<<dim3(DMODEL / 64, M16 / 256), 256, 0, stream>>>(
      hid1B, h_w2T, h_b2, nullptr, hF32, M16, DMODEL, DMODEL, 1);

  // ---- 4. cls token + pos_emb ----
  {
    int total = M17 * DMODEL;
    assemble_kernel<<<ceil_div(total, 256), 256, 0, stream>>>(hF32, cls_tok, pos_emb, total);
  }

  // ---- 5. encoder layers ----
  for (int l = 0; l < NLAYER; ++l) {
    ln_kernel<<<M17 / 8, 256, 0, stream>>>(hF32, ln1_s + l * DMODEL, ln1_b + l * DMODEL,
                                           nb, M17, 0);
    gemm_bf16_kernel<EPI_F32><<<dim3(3 * DMODEL / 64, M17 / 256), 256, 0, stream>>>(
        nb, qkv_wT + (size_t)l * 3 * DMODEL * DMODEL, qkv_b + (size_t)l * 3 * DMODEL,
        nullptr, qkvF, M17, 3 * DMODEL, DMODEL, 0);
    attn_kernel<<<(BATCH * NHEAD) / 8, 256, 0, stream>>>(qkvF, nb);
    gemm_bf16_kernel<EPI_RESID_F32><<<dim3(DMODEL / 64, M17 / 256), 256, 0, stream>>>(
        nb, proj_wT + (size_t)l * DMODEL * DMODEL, proj_b + (size_t)l * DMODEL,
        hF32, hF32, M17, DMODEL, DMODEL, 0);
    ln_kernel<<<M17 / 8, 256, 0, stream>>>(hF32, ln2_s + l * DMODEL, ln2_b + l * DMODEL,
                                           nb, M17, 0);
    gemm_bf16_kernel<EPI_GELU_BF16><<<dim3(DFF / 64, M17 / 256), 256, 0, stream>>>(
        nb, mlp_w1T + (size_t)l * DFF * DMODEL, mlp_b1 + (size_t)l * DFF,
        nullptr, hidB, M17, DFF, DMODEL, 0);
    gemm_bf16_kernel<EPI_RESID_F32><<<dim3(DMODEL / 64, M17 / 256), 256, 0, stream>>>(
        hidB, mlp_w2T + (size_t)l * DMODEL * DFF, mlp_b2 + (size_t)l * DMODEL,
        hF32, hF32, M17, DMODEL, DFF, 0);
  }

  // ---- 6. final LN on cls rows -> nb (2048 x 512 bf16) ----
  ln_kernel<<<BATCH / 8, 256, 0, stream>>>(hF32, fin_s, fin_b, nb, BATCH, 1);

  // ---- 7. classifier: (2048 x 10 x 512) -> d_out f32 ----
  gemm_bf16_kernel<EPI_F32><<<dim3(1, BATCH / 256), 256, 0, stream>>>(
      nb, cls_wT, cls_b, nullptr, d_out, BATCH, NCLS, DMODEL, 0);
}
